// GCNNet_16552803958871
// MI455X (gfx1250) — matrix operand order repaired and checked by reading, not compiled, not run on hardware
//
#include <hip/hip_runtime.h>
#include <hip/hip_bf16.h>
#include <math.h>

// ---------------------------------------------------------------------------
// Types for CDNA5 WMMA (gfx1250, wave32)
// ---------------------------------------------------------------------------
typedef __attribute__((ext_vector_type(16))) __bf16 v16bf;
typedef __attribute__((ext_vector_type(8)))  __bf16 v8bf;
typedef __attribute__((ext_vector_type(8)))  float  v8f;

// ---------------------------------------------------------------------------
// Small utility kernels
// ---------------------------------------------------------------------------
__global__ void fill_kernel(float* __restrict__ p, float v, long long n) {
    long long i = (long long)blockIdx.x * blockDim.x + threadIdx.x;
    if (i < n) p[i] = v;
}

__global__ void deg_kernel(const int* __restrict__ dst, float* __restrict__ deg,
                           long long E) {
    long long e = (long long)blockIdx.x * blockDim.x + threadIdx.x;
    if (e < E) atomicAdd(&deg[dst[e]], 1.0f);
}

__global__ void rsqrt_kernel(float* __restrict__ p, long long n) {
    long long i = (long long)blockIdx.x * blockDim.x + threadIdx.x;
    if (i < n) p[i] = rsqrtf(p[i]);
}

// ---------------------------------------------------------------------------
// WMMA GEMM: H[n, DOUT] = X[n, DIN] @ W[DIN, DOUT]   (bf16 in, f32 accumulate)
//
// Block = 128 threads = 4 waves; wave w owns rows [block*64 + w*16, +16).
// Each wave keeps NT accumulator fragments and sweeps all column tiles,
// reusing one A fragment per K-step (A traffic amortized over all of DOUT).
//
// W is staged once per block into LDS as zero-padded bf16, column-major
// with a padded K-stride, so B fragments are branch-free ds_load_b128 pairs.
//
// Fragment layouts used here (contiguous K order, shared by A and B):
//   A (16x32 bf16): lane<16 -> row=lane,    elems = K0..15
//                   lane>=16 -> row=lane-16, elems = K16..31
//   B (32x16 bf16): lane<16 -> col=lane,    elems = K0..15
//                   lane>=16 -> col=lane-16, elems = K16..31
//   C/D (16x16 f32): vgpr r -> row = r (+8 for lanes>=16), col = lane&15
// ---------------------------------------------------------------------------
template <int DIN, int DOUT>
__global__ void gemm_wmma_kernel(const float* __restrict__ X,
                                 const float* __restrict__ W,
                                 float* __restrict__ H, int n) {
    constexpr int NT   = (DOUT + 15) / 16;   // 16-col tiles
    constexpr int NP   = NT * 16;            // padded column count
    constexpr int DINP = DIN + 8;            // padded K stride (bank spread)

    __shared__ __bf16 sW[NP * DINP];         // column-major: sW[c*DINP + k]

    // ---- stage W -> LDS (bf16, zero-padded columns) ----
    for (int idx = threadIdx.x; idx < NP * DIN; idx += blockDim.x) {
        const int c = idx / DIN;
        const int k = idx - c * DIN;
        const float w = (c < DOUT) ? W[(long long)k * DOUT + c] : 0.0f;
        sW[c * DINP + k] = (__bf16)w;
    }
    __syncthreads();

    const int wave = threadIdx.x >> 5;       // 0..3 : 16-row strip within block
    const int lane = threadIdx.x & 31;
    const int half = lane >> 4;              // 0: lanes 0-15, 1: lanes 16-31
    const int l16  = lane & 15;
    const int m_base = blockIdx.x * 64 + wave * 16;

    int rowA = m_base + l16;
    if (rowA >= n) rowA = n - 1;             // clamp; stores are masked below
    const float* __restrict__ xrow = X + (long long)rowA * DIN;

    v8f acc[NT];
#pragma unroll
    for (int t = 0; t < NT; ++t) acc[t] = (v8f){};

#pragma unroll
    for (int k0 = 0; k0 < DIN; k0 += 32) {
        // ---- A fragment (global, contiguous per lane) ----
        v16bf a;
        const int kA = k0 + half * 16;
#pragma unroll
        for (int i = 0; i < 8; ++i) {
            a[i]     = (__bf16)xrow[kA + i];
            a[i + 8] = (__bf16)xrow[kA + 8 + i];
        }
        const int kB = k0 + half * 16;       // 16 contiguous K per lane
#pragma unroll
        for (int t = 0; t < NT; ++t) {
            // ---- B fragment: two 16B LDS vector loads, no branches ----
            const __bf16* bp = &sW[(t * 16 + l16) * DINP + kB];
            const v8bf blo = *(const v8bf*)bp;
            const v8bf bhi = *(const v8bf*)(bp + 8);
            v16bf b;
#pragma unroll
            for (int i = 0; i < 8; ++i) { b[i] = blo[i]; b[i + 8] = bhi[i]; }
            // D = A*B + C   -> v_wmma_f32_16x16x32_bf16
            acc[t] = __builtin_amdgcn_wmma_f32_16x16x32_bf16(
                /*neg_a=*/false, a, /*neg_b=*/false, b,
                /*c_mod=*/(short)0, acc[t], /*reuse_a=*/false, /*reuse_b=*/false);
        }
    }

    // ---- store D tiles ----
#pragma unroll
    for (int t = 0; t < NT; ++t) {
        const int col = t * 16 + l16;
#pragma unroll
        for (int r = 0; r < 8; ++r) {
            const int row = m_base + r + half * 8;
            if (row < n && col < DOUT) H[(long long)row * DOUT + col] = acc[t][r];
        }
    }
}

// ---------------------------------------------------------------------------
// Edge scatter: acc[dst] += dinv[src] * H[src]   (per feature, f32 atomics)
// Each thread handles V features of one edge (vector load from H).
// ---------------------------------------------------------------------------
template <int F, int V>
__global__ void edge_scatter_kernel(const int* __restrict__ src,
                                    const int* __restrict__ dst,
                                    const float* __restrict__ dinv,
                                    const float* __restrict__ H,
                                    float* __restrict__ acc, long long E) {
    constexpr int C = F / V;                 // chunks per edge
    long long idx = (long long)blockIdx.x * blockDim.x + threadIdx.x;
    const long long total = E * C;
    if (idx >= total) return;
    const long long e = idx / C;
    const int c = (int)(idx - e * C);
    const int s = src[e];
    const int d = dst[e];
    const float w = dinv[s];
    const float* __restrict__ hp = H + (long long)s * F + (long long)c * V;
    float* __restrict__ ap = acc + (long long)d * F + (long long)c * V;
#pragma unroll
    for (int i = 0; i < V; ++i) atomicAdd(ap + i, w * hp[i]);
}

// ---------------------------------------------------------------------------
// Finalize: out = [relu]( dinv[i]*(acc[i,f] + dinv[i]*H[i,f]) + b[f] )
// ---------------------------------------------------------------------------
__global__ void finalize_kernel(const float* __restrict__ acc,
                                const float* __restrict__ H,
                                const float* __restrict__ dinv,
                                const float* __restrict__ bias,
                                float* __restrict__ out,
                                long long n, int F, int do_relu) {
    long long idx = (long long)blockIdx.x * blockDim.x + threadIdx.x;
    const long long total = n * F;
    if (idx >= total) return;
    const long long i = idx / F;
    const int f = (int)(idx - i * F);
    const float di = dinv[i];
    float v = di * (acc[idx] + di * H[idx]) + bias[f];
    if (do_relu) v = fmaxf(v, 0.0f);
    out[idx] = v;
}

// ---------------------------------------------------------------------------
// Last layer: finalize (no relu) + log_softmax over 6 logits per node
// ---------------------------------------------------------------------------
__global__ void finalize_lsm6_kernel(const float* __restrict__ acc,
                                     const float* __restrict__ H,
                                     const float* __restrict__ dinv,
                                     const float* __restrict__ bias,
                                     float* __restrict__ out, long long n) {
    long long i = (long long)blockIdx.x * blockDim.x + threadIdx.x;
    if (i >= n) return;
    const float di = dinv[i];
    float z[6];
    float m = -3.402823466e+38f;
#pragma unroll
    for (int f = 0; f < 6; ++f) {
        const long long idx = i * 6 + f;
        z[f] = di * (acc[idx] + di * H[idx]) + bias[f];
        m = fmaxf(m, z[f]);
    }
    float s = 0.0f;
#pragma unroll
    for (int f = 0; f < 6; ++f) s += __expf(z[f] - m);
    const float l = __logf(s);
#pragma unroll
    for (int f = 0; f < 6; ++f) out[i * 6 + f] = z[f] - m - l;
}

// ---------------------------------------------------------------------------
// Launcher
// ---------------------------------------------------------------------------
static inline unsigned nblk(long long total, int b) {
    return (unsigned)((total + b - 1) / b);
}

extern "C" void kernel_launch(void* const* d_in, const int* in_sizes, int n_in,
                              void* d_out, int out_size, void* d_ws, size_t ws_size,
                              hipStream_t stream) {
    const float* x  = (const float*)d_in[0];
    const int*   ei = (const int*)d_in[1];       // [2, E] int32 (JAX default x64-off)
    const float* W1 = (const float*)d_in[2];
    const float* b1 = (const float*)d_in[3];
    const float* W2 = (const float*)d_in[4];
    const float* b2 = (const float*)d_in[5];
    const float* W3 = (const float*)d_in[6];
    const float* b3 = (const float*)d_in[7];

    const long long n = in_sizes[0] / 128;       // N nodes
    const long long E = in_sizes[1] / 2;         // edges
    const int* src = ei;
    const int* dst = ei + E;

    // Workspace layout (floats): dinv[n] | H[n*64] | acc[n*64] | Xb[n*64]
    float* ws   = (float*)d_ws;
    float* dinv = ws;
    float* H    = dinv + n;
    float* acc  = H + n * 64;
    float* Xb   = acc + n * 64;

    const int T = 256;

    // ---- degrees -> dinv (self-loop gives deg >= 1) ----
    fill_kernel<<<nblk(n, T), T, 0, stream>>>(dinv, 1.0f, n);
    deg_kernel<<<nblk(E, T), T, 0, stream>>>(dst, dinv, E);
    rsqrt_kernel<<<nblk(n, T), T, 0, stream>>>(dinv, n);

    const unsigned gemm_blocks = nblk(n, 64);    // 64 rows per block (4 waves)

    // ---- layer 1: 128 -> 64, relu ----
    gemm_wmma_kernel<128, 64><<<gemm_blocks, 128, 0, stream>>>(x, W1, H, (int)n);
    fill_kernel<<<nblk(n * 64, T), T, 0, stream>>>(acc, 0.0f, n * 64);
    edge_scatter_kernel<64, 4><<<nblk(E * 16, T), T, 0, stream>>>(src, dst, dinv, H, acc, E);
    finalize_kernel<<<nblk(n * 64, T), T, 0, stream>>>(acc, H, dinv, b1, Xb, n, 64, 1);

    // ---- layer 2: 64 -> 64, relu ----
    gemm_wmma_kernel<64, 64><<<gemm_blocks, 128, 0, stream>>>(Xb, W2, H, (int)n);
    fill_kernel<<<nblk(n * 64, T), T, 0, stream>>>(acc, 0.0f, n * 64);
    edge_scatter_kernel<64, 4><<<nblk(E * 16, T), T, 0, stream>>>(src, dst, dinv, H, acc, E);
    finalize_kernel<<<nblk(n * 64, T), T, 0, stream>>>(acc, H, dinv, b2, Xb, n, 64, 1);

    // ---- layer 3: 64 -> 6, log_softmax ----
    gemm_wmma_kernel<64, 6><<<gemm_blocks, 128, 0, stream>>>(Xb, W3, H, (int)n);
    fill_kernel<<<nblk(n * 6, T), T, 0, stream>>>(acc, 0.0f, n * 6);
    edge_scatter_kernel<6, 2><<<nblk(E * 3, T), T, 0, stream>>>(src, dst, dinv, H, acc, E);
    finalize_lsm6_kernel<<<nblk(n, T), T, 0, stream>>>(acc, H, dinv, b3, (float*)d_out, n);
}
